// MentionPruner_22497038696559
// MI455X (gfx1250) — matrix operand order repaired and checked by reading, not compiled, not run on hardware
//
#include <hip/hip_runtime.h>
#include <hip/hip_bf16.h>

#define DEV __device__ __forceinline__

typedef __attribute__((ext_vector_type(16))) __bf16 v16bf;
typedef __attribute__((ext_vector_type(8)))  float  v8f;

// ---------------- problem constants ----------------
constexpr int Bc    = 8;
constexpr int Nc    = 10240;        // SEQ * MAX_SPAN_LENGTH
constexpr int Dc    = 2048;
constexpr int Hc    = 512;
constexpr int Kc    = 410;          // kept spans
constexpr int MSPAN = 5;
constexpr int Mrows = Bc * Nc;      // 81920 total rows
constexpr int MT    = 128;          // rows per scorer block

// ---------------- d_out flat layout (reference tuple order) ----------------
constexpr size_t OFF_SCORES = 0;                                   // [B,N,1]
constexpr size_t OFF_IDX    = OFF_SCORES + (size_t)Bc * Nc;        // [B,K]
constexpr size_t OFF_PVEC   = OFF_IDX    + (size_t)Bc * Kc;        // [B,K,D]
constexpr size_t OFF_PSC    = OFF_PVEC   + (size_t)Bc * Kc * Dc;   // [B,K,1]
constexpr size_t OFF_PBEG   = OFF_PSC    + (size_t)Bc * Kc;        // [B,K,1]
constexpr size_t OFF_PEND   = OFF_PBEG   + (size_t)Bc * Kc;        // [B,K,1]
constexpr size_t OFF_SLEN   = OFF_PEND   + (size_t)Bc * Kc;        // [B]
constexpr size_t OFF_SQ     = OFF_SLEN   + (size_t)Bc;             // [B,K,K]
constexpr size_t OFF_TRI    = OFF_SQ     + (size_t)Bc * Kc * Kc;   // [B,K,K]
constexpr size_t OFF_SPANS  = OFF_TRI    + (size_t)Bc * Kc * Kc;   // [B,K,2]

// ---------------- helpers ----------------
DEV unsigned short f2bf(float f) {
    unsigned int u = __float_as_uint(f);
    unsigned int r = u + 0x7FFFu + ((u >> 16) & 1u);   // round-to-nearest-even
    return (unsigned short)(r >> 16);
}

union BF16x16 {
    v16bf v;
    unsigned short u[16];
    uint4 q[2];
};

DEV v8f v8f_zero() {
    v8f z;
#pragma unroll
    for (int i = 0; i < 8; ++i) z[i] = 0.0f;
    return z;
}

// A operand (16x32 bf16, M x K) built from row-major fp32 source (fallback path).
// Per ISA layout: lane m = lane&15, kb = 8*(lane>>4);
// vector elems 0..7 -> K = kb..kb+7 ; elems 8..15 -> K = 16+kb..16+kb+7
DEV v16bf load_a_f32(const float* src, int stride, int m, int k0, int lane) {
    int kb = (lane >> 4) << 3;
    const float* p0 = src + (size_t)m * stride + k0 + kb;
    const float* p1 = p0 + 16;
    BF16x16 r;
#pragma unroll
    for (int e = 0; e < 8; ++e) {
        r.u[e]     = f2bf(p0[e]);
        r.u[8 + e] = f2bf(p1[e]);
    }
    return r.v;
}

// A operand from row-major bf16 source: two straight b128 loads, no VALU.
DEV v16bf load_a_bf16(const unsigned short* src, int stride, int m, int k0, int lane) {
    int kb = (lane >> 4) << 3;
    const unsigned short* p0 = src + (size_t)m * stride + k0 + kb;
    BF16x16 r;
    r.q[0] = *(const uint4*)(p0);        // K = kb..kb+7
    r.q[1] = *(const uint4*)(p0 + 16);   // K = 16+kb..16+kb+7
    return r.v;
}

// Same A layout from an LDS bf16 (ushort) row-major tile: two b128 ds reads.
DEV v16bf load_a_lds(const unsigned short* h, int stride, int m, int k0, int lane) {
    int kb = (lane >> 4) << 3;
    const unsigned short* p0 = h + (size_t)m * stride + k0 + kb;
    BF16x16 r;
    r.q[0] = *(const uint4*)(p0);
    r.q[1] = *(const uint4*)(p0 + 16);
    return r.v;
}

// B operand (32x16 bf16, K x N) from the packed weight buffer:
// packed index = ((kt*NT + nt)*32 + lane)*16 + e  -> 32 contiguous bytes per lane.
DEV v16bf load_b_packed(const unsigned short* wp, int kt, int nt, int NT, int lane) {
    const unsigned short* p = wp + ((((size_t)kt * NT + nt) * 32) + lane) * 16;
    BF16x16 r;
    r.q[0] = *(const uint4*)(p);
    r.q[1] = *(const uint4*)(p + 8);
    return r.v;
}

// ---------------- kernel 0a: pack fp32 weights into WMMA B-operand bf16 layout ----
__global__ __launch_bounds__(256) void pack_weights(const float* __restrict__ W,
                                                    unsigned short* __restrict__ Wp,
                                                    int Kdim, int Ncol) {
    int tid = blockIdx.x * blockDim.x + threadIdx.x;
    int total = Kdim * Ncol;
    if (tid >= total) return;
    int e    = tid & 15;
    int lane = (tid >> 4) & 31;
    int t    = tid >> 9;                 // kt*NT + nt
    int NT   = Ncol >> 4;
    int nt   = t % NT;
    int kt   = t / NT;
    int nn   = lane & 15;
    int kk   = ((lane >> 4) << 3) + (e & 7) + ((e >> 3) << 4);   // B layout: elems 0-7 K=kb..kb+7, elems 8-15 K=16+kb..16+kb+7, kb=8*(lane>>4)
    Wp[tid] = f2bf(W[(size_t)(kt * 32 + kk) * Ncol + (nt * 16 + nn)]);
}

// ---------------- kernel 0b: streaming fp32 -> bf16 conversion of X ----------------
// 8 elements per thread: two float4 loads -> one uint4 store.
__global__ __launch_bounds__(256) void convert_x(const float* __restrict__ X,
                                                 unsigned short* __restrict__ Xbf,
                                                 size_t n8) {
    size_t t = (size_t)blockIdx.x * blockDim.x + threadIdx.x;
    if (t >= n8) return;
    const float4* p = (const float4*)(X + t * 8);
    float4 a = p[0], b = p[1];
    BF16x16 r;   // only low 8 used
    r.u[0] = f2bf(a.x); r.u[1] = f2bf(a.y); r.u[2] = f2bf(a.z); r.u[3] = f2bf(a.w);
    r.u[4] = f2bf(b.x); r.u[5] = f2bf(b.y); r.u[6] = f2bf(b.z); r.u[7] = f2bf(b.w);
    *(uint4*)(Xbf + t * 8) = r.q[0];
}

// ---------------- kernel 1: fused scorer MLP (WMMA bf16, fp32 accum) ----------------
// block = 256 threads (8 waves); block owns 128 rows of X=[81920 x 2048].
// USE_BF: A operand comes from pre-converted bf16 X (zero VALU in hot loop).
template <bool USE_BF>
__global__ __launch_bounds__(256) void scorer_kernel(
    const float* __restrict__ X, const unsigned short* __restrict__ Xbf,
    const float* __restrict__ span_mask,
    const unsigned short* __restrict__ W1p, const float* __restrict__ b1,
    const unsigned short* __restrict__ W2p, const float* __restrict__ b2,
    const float* __restrict__ W3, const float* __restrict__ b3,
    float* __restrict__ scores_out) {
    extern __shared__ char dynsm[];
    unsigned short* h1 = (unsigned short*)dynsm;               // 128 x 512 bf16 (128 KB)
    float* sscore2 = (float*)(dynsm + (size_t)MT * Hc * 2);    // 128 x 2 partial row dots

    const int lane = threadIdx.x & 31;
    const int wave = threadIdx.x >> 5;
    const int wr   = wave & 3;   // rows  32*wr .. 32*wr+31
    const int wc   = wave >> 2;  // cols  64*wc .. 64*wc+63 within a 128-col chunk
    const size_t rowBase = (size_t)blockIdx.x * MT;
    const float* Xblk = X + rowBase * Dc;
    const unsigned short* Xbfblk = Xbf + rowBase * Dc;

    // ---- GEMM1: h1 = relu(X*W1 + b1), bf16 into LDS. 4 column chunks of 128. ----
#pragma unroll 1
    for (int cc = 0; cc < 4; ++cc) {
        v8f acc[2][4];
#pragma unroll
        for (int rt = 0; rt < 2; ++rt)
#pragma unroll
            for (int ct = 0; ct < 4; ++ct) acc[rt][ct] = v8f_zero();

        for (int kt = 0; kt < Dc / 32; ++kt) {
            v16bf a[2];
            if (USE_BF) {
#pragma unroll
                for (int rt = 0; rt < 2; ++rt)
                    a[rt] = load_a_bf16(Xbfblk, Dc, wr * 32 + rt * 16 + (lane & 15), kt * 32, lane);
            } else {
                __builtin_prefetch(Xblk + ((size_t)(wr * 32 + (lane & 15))) * Dc + (kt + 1) * 32, 0, 1);
#pragma unroll
                for (int rt = 0; rt < 2; ++rt)
                    a[rt] = load_a_f32(Xblk, Dc, wr * 32 + rt * 16 + (lane & 15), kt * 32, lane);
            }
            v16bf bop[4];
#pragma unroll
            for (int ct = 0; ct < 4; ++ct) {
                int nt = cc * 8 + wc * 4 + ct;   // of 32 column tiles
                bop[ct] = load_b_packed(W1p, kt, nt, Hc >> 4, lane);
            }
#pragma unroll
            for (int rt = 0; rt < 2; ++rt)
#pragma unroll
                for (int ct = 0; ct < 4; ++ct)
                    acc[rt][ct] = __builtin_amdgcn_wmma_f32_16x16x32_bf16(
                        false, a[rt], false, bop[ct], (short)0, acc[rt][ct], false, false);
        }
        // epilogue: bias + relu -> bf16 LDS
#pragma unroll
        for (int ct = 0; ct < 4; ++ct) {
            int col = cc * 128 + wc * 64 + ct * 16 + (lane & 15);
            float bb = b1[col];
#pragma unroll
            for (int rt = 0; rt < 2; ++rt) {
                int rb = wr * 32 + rt * 16 + ((lane >> 4) << 3);
#pragma unroll
                for (int j = 0; j < 8; ++j) {
                    float v = acc[rt][ct][j] + bb;
                    v = v > 0.0f ? v : 0.0f;
                    h1[(size_t)(rb + j) * Hc + col] = f2bf(v);
                }
            }
        }
    }
    __syncthreads();

    // ---- GEMM2 + fused GEMV: score = relu(h1*W2 + b2) . W3 ----
    float partial[2][8];
#pragma unroll
    for (int rt = 0; rt < 2; ++rt)
#pragma unroll
        for (int j = 0; j < 8; ++j) partial[rt][j] = 0.0f;

#pragma unroll 1
    for (int cc = 0; cc < 4; ++cc) {
        v8f acc[2][4];
#pragma unroll
        for (int rt = 0; rt < 2; ++rt)
#pragma unroll
            for (int ct = 0; ct < 4; ++ct) acc[rt][ct] = v8f_zero();

        for (int kt = 0; kt < Hc / 32; ++kt) {
            v16bf a[2];
#pragma unroll
            for (int rt = 0; rt < 2; ++rt)
                a[rt] = load_a_lds(h1, Hc, wr * 32 + rt * 16 + (lane & 15), kt * 32, lane);
            v16bf bop[4];
#pragma unroll
            for (int ct = 0; ct < 4; ++ct) {
                int nt = cc * 8 + wc * 4 + ct;
                bop[ct] = load_b_packed(W2p, kt, nt, Hc >> 4, lane);
            }
#pragma unroll
            for (int rt = 0; rt < 2; ++rt)
#pragma unroll
                for (int ct = 0; ct < 4; ++ct)
                    acc[rt][ct] = __builtin_amdgcn_wmma_f32_16x16x32_bf16(
                        false, a[rt], false, bop[ct], (short)0, acc[rt][ct], false, false);
        }
        // epilogue: relu(h2) * W3 accumulated per lane (deterministic)
#pragma unroll
        for (int ct = 0; ct < 4; ++ct) {
            int col = cc * 128 + wc * 64 + ct * 16 + (lane & 15);
            float bb = b2[col];
            float w3 = W3[col];
#pragma unroll
            for (int rt = 0; rt < 2; ++rt)
#pragma unroll
                for (int j = 0; j < 8; ++j) {
                    float v = acc[rt][ct][j] + bb;
                    v = v > 0.0f ? v : 0.0f;
                    partial[rt][j] += v * w3;
                }
        }
    }
    // cross-lane (16 columns) deterministic tree reduction
#pragma unroll
    for (int rt = 0; rt < 2; ++rt)
#pragma unroll
        for (int j = 0; j < 8; ++j) {
            float s = partial[rt][j];
            s += __shfl_xor(s, 1, 32);
            s += __shfl_xor(s, 2, 32);
            s += __shfl_xor(s, 4, 32);
            s += __shfl_xor(s, 8, 32);
            if ((lane & 15) == 0) {
                int row = wr * 32 + rt * 16 + ((lane >> 4) << 3) + j;
                sscore2[row * 2 + wc] = s;   // unique (row, wc) writer
            }
        }
    __syncthreads();

    if (threadIdx.x < MT) {
        size_t r = rowBase + threadIdx.x;
        float s = sscore2[threadIdx.x * 2] + sscore2[threadIdx.x * 2 + 1] + b3[0];
        s -= (1.0f - span_mask[r]) * 10000.0f;
        scores_out[r] = s;
    }
}

// ---------------- kernel 2: per-batch top-K via LDS bitonic sort ----------------
__global__ __launch_bounds__(1024) void topk_kernel(const float* __restrict__ scores,
                                                    const int* __restrict__ seqlen,
                                                    int* __restrict__ idx_ws,
                                                    int* __restrict__ slen_ws,
                                                    float* __restrict__ out) {
    extern __shared__ char dynsm[];
    unsigned long long* keys = (unsigned long long*)dynsm;   // 16384 x 8B = 128 KB
    const int NPAD = 16384;
    const int b = blockIdx.x;
    const int t = threadIdx.x;

    // pack (orderable-float << 32) | tie-break(lower index wins) ; pad with 0
    for (int i = t; i < NPAD; i += 1024) {
        unsigned long long kk = 0ull;
        if (i < Nc) {
            unsigned int u = __float_as_uint(scores[(size_t)b * Nc + i]);
            u = (u & 0x80000000u) ? ~u : (u | 0x80000000u);
            kk = ((unsigned long long)u << 32) | (unsigned int)(0x7FFFFFFF - i);
        }
        keys[i] = kk;
    }
    __syncthreads();

    // bitonic sort, descending
    for (int k = 2; k <= NPAD; k <<= 1) {
        for (int j = k >> 1; j > 0; j >>= 1) {
            for (int i = t; i < NPAD; i += 1024) {
                int ixj = i ^ j;
                if (ixj > i) {
                    unsigned long long A = keys[i], Bv = keys[ixj];
                    bool segUp = ((i & k) == 0);
                    bool doSwap = segUp ? (A < Bv) : (A > Bv);
                    if (doSwap) { keys[i] = Bv; keys[ixj] = A; }
                }
            }
            __syncthreads();
        }
    }

    const int slen = (int)((float)seqlen[b] * 0.2f) + 1;     // <= Kc always

    // extract kept indices; invalid slots -> N-1; pad to 512 with N-1
    int myv = Nc - 1;
    if (t < 512) {
        if (t < Kc) {
            int n = 0x7FFFFFFF - (int)(unsigned int)(keys[t] & 0xFFFFFFFFu);
            myv = (t < slen) ? n : (Nc - 1);
        }
    }
    __syncthreads();
    int* si = (int*)dynsm;   // reuse LDS
    if (t < 512) si[t] = myv;
    __syncthreads();

    // bitonic sort 512 ints, ascending
    for (int k = 2; k <= 512; k <<= 1) {
        for (int j = k >> 1; j > 0; j >>= 1) {
            for (int i = t; i < 512; i += 1024) {
                int ixj = i ^ j;
                if (ixj > i) {
                    int A = si[i], Bv = si[ixj];
                    bool segUp = ((i & k) == 0);
                    bool doSwap = segUp ? (A > Bv) : (A < Bv);
                    if (doSwap) { si[i] = Bv; si[ixj] = A; }
                }
            }
            __syncthreads();
        }
    }

    if (t < Kc) {
        int v = si[t];
        idx_ws[b * Kc + t] = v;
        out[OFF_IDX + (size_t)b * Kc + t] = (float)v;
    }
    if (t == 0) {
        slen_ws[b] = slen;
        out[OFF_SLEN + b] = (float)slen;
    }
}

// ---------------- kernel 3: gathers, masks, spans ----------------
__global__ __launch_bounds__(256) void outputs_kernel(const float* __restrict__ span_vecs,
                                                      const int* __restrict__ span_begin,
                                                      const int* __restrict__ span_end,
                                                      const int* __restrict__ idx_ws,
                                                      const int* __restrict__ slen_ws,
                                                      float* __restrict__ out) {
    const float* scores = out + OFF_SCORES;
    const size_t NV  = (size_t)Bc * Kc * Dc;
    const size_t BK  = (size_t)Bc * Kc;
    const size_t BKK = (size_t)Bc * Kc * Kc;
    const size_t total = NV + 3 * BK + 2 * BKK + BK * 2;

    for (size_t t = (size_t)blockIdx.x * blockDim.x + threadIdx.x; t < total;
         t += (size_t)gridDim.x * blockDim.x) {
        size_t r = t;
        if (r < NV) {   // pruned_vecs
            size_t b = r / ((size_t)Kc * Dc);
            size_t rem = r % ((size_t)Kc * Dc);
            size_t k = rem / Dc, d = rem % Dc;
            int id = idx_ws[b * Kc + k];
            out[OFF_PVEC + r] = span_vecs[(b * (size_t)Nc + id) * Dc + d];
            continue;
        }
        r -= NV;
        if (r < BK) {   // pruned_scores
            size_t b = r / Kc;
            int id = idx_ws[r];
            out[OFF_PSC + r] = scores[b * Nc + id];
            continue;
        }
        r -= BK;
        if (r < BK) {   // pruned_begin
            size_t b = r / Kc;
            int id = idx_ws[r];
            out[OFF_PBEG + r] = (float)span_begin[b * Nc + id];
            continue;
        }
        r -= BK;
        if (r < BK) {   // pruned_end
            size_t b = r / Kc;
            int id = idx_ws[r];
            out[OFF_PEND + r] = (float)span_end[b * Nc + id];
            continue;
        }
        r -= BK;
        if (r < BKK) {  // square_mask
            size_t b = r / ((size_t)Kc * Kc);
            size_t rem = r % ((size_t)Kc * Kc);
            size_t i = rem / Kc, j = rem % Kc;
            int L = slen_ws[b];
            out[OFF_SQ + r] = (i < (size_t)L && j < (size_t)L) ? 1.0f : 0.0f;
            continue;
        }
        r -= BKK;
        if (r < BKK) {  // triangular_mask
            size_t b = r / ((size_t)Kc * Kc);
            size_t rem = r % ((size_t)Kc * Kc);
            size_t i = rem / Kc, j = rem % Kc;
            int L = slen_ws[b];
            out[OFF_TRI + r] = (i < (size_t)L && j < (size_t)L && j <= i) ? 1.0f : 0.0f;
            continue;
        }
        r -= BKK;
        {               // spans
            size_t b = r / ((size_t)Kc * 2);
            size_t rem = r % ((size_t)Kc * 2);
            size_t k = rem >> 1;
            int c = (int)(rem & 1);
            int id = idx_ws[b * Kc + k];
            int sb = id / MSPAN;
            out[OFF_SPANS + r] = (float)(c == 0 ? sb : sb + id % MSPAN);
        }
    }
}

// ---------------- host entry ----------------
extern "C" void kernel_launch(void* const* d_in, const int* in_sizes, int n_in,
                              void* d_out, int out_size, void* d_ws, size_t ws_size,
                              hipStream_t stream) {
    const float* span_vecs  = (const float*)d_in[0];
    const float* span_mask  = (const float*)d_in[1];
    const int*   span_begin = (const int*)d_in[2];
    const int*   span_end   = (const int*)d_in[3];
    const int*   seqlen     = (const int*)d_in[4];
    const float* W1 = (const float*)d_in[5];
    const float* b1 = (const float*)d_in[6];
    const float* W2 = (const float*)d_in[7];
    const float* b2 = (const float*)d_in[8];
    const float* W3 = (const float*)d_in[9];
    const float* b3 = (const float*)d_in[10];
    float* out = (float*)d_out;

    const size_t xbf_elems = (size_t)Mrows * Dc;                       // 167,772,160
    const size_t w_elems   = (size_t)Dc * Hc + (size_t)Hc * Hc;        // packed weights
    const size_t small_bytes = w_elems * 2 + ((size_t)Bc * Kc + Bc) * 4 + 256;
    const bool useXbf = ws_size >= xbf_elems * 2 + small_bytes;

    unsigned short* Xbf = nullptr;
    unsigned short* W1p;
    if (useXbf) {
        Xbf = (unsigned short*)d_ws;
        W1p = Xbf + xbf_elems;
    } else {
        W1p = (unsigned short*)d_ws;
    }
    unsigned short* W2p = W1p + (size_t)Dc * Hc;
    int* idx_ws  = (int*)(W2p + (size_t)Hc * Hc);
    int* slen_ws = idx_ws + Bc * Kc;

    pack_weights<<<(Dc * Hc + 255) / 256, 256, 0, stream>>>(W1, W1p, Dc, Hc);
    pack_weights<<<(Hc * Hc + 255) / 256, 256, 0, stream>>>(W2, W2p, Hc, Hc);

    size_t smem1 = (size_t)MT * Hc * 2 + (size_t)MT * 2 * sizeof(float);
    if (useXbf) {
        size_t n8 = xbf_elems / 8;
        convert_x<<<(unsigned)((n8 + 255) / 256), 256, 0, stream>>>(span_vecs, Xbf, n8);
        scorer_kernel<true><<<Mrows / MT, 256, smem1, stream>>>(
            span_vecs, Xbf, span_mask, W1p, b1, W2p, b2, W3, b3, out + OFF_SCORES);
    } else {
        scorer_kernel<false><<<Mrows / MT, 256, smem1, stream>>>(
            span_vecs, Xbf, span_mask, W1p, b1, W2p, b2, W3, b3, out + OFF_SCORES);
    }

    topk_kernel<<<Bc, 1024, (size_t)16384 * 8, stream>>>(
        out + OFF_SCORES, seqlen, idx_ws, slen_ws, out);

    const size_t NV  = (size_t)Bc * Kc * Dc;
    const size_t BK  = (size_t)Bc * Kc;
    const size_t BKK = (size_t)Bc * Kc * Kc;
    const size_t total = NV + 3 * BK + 2 * BKK + BK * 2;
    int blocks = (int)((total + 255) / 256);
    outputs_kernel<<<blocks, 256, 0, stream>>>(span_vecs, span_begin, span_end,
                                               idx_ws, slen_ws, out);
}
